// FusedSparseConvLIF_88064009437932
// MI455X (gfx1250) — matrix operand order repaired and checked by reading, not compiled, not run on hardware
//
#include <hip/hip_runtime.h>

// ---------------------------------------------------------------------------
// Fused 3x3 conv (implicit GEMM via v_wmma_f32_16x16x32_f16) + LIF dynamics.
// v5: 4-row spatial tile, 512-thread block (16 waves) -> halo amplification
// 1.5x (HBM floor ~58us @ 23.3 TB/s). Weight fragment table staged
// global->LDS via the Tensor Data Mover (tensor_load_to_lds, 6-arg builtin
// per this toolchain, + s_wait_tensorcnt). Membrane state v stays in VGPRs
// across all T=4 timesteps.
// ---------------------------------------------------------------------------

typedef __attribute__((ext_vector_type(16))) _Float16 v16h;
typedef __attribute__((ext_vector_type(8)))  _Float16 v8h;
typedef __attribute__((ext_vector_type(8)))  float    v8f;
typedef __attribute__((ext_vector_type(4)))  unsigned int u32x4;
typedef __attribute__((ext_vector_type(8)))  int          i32x8;
typedef __attribute__((ext_vector_type(4)))  int          i32x4;

constexpr int T_   = 4;
constexpr int B_   = 8;
constexpr int CIN  = 64;
constexpr int COUT = 64;
constexpr int H_   = 128;
constexpr int W_   = 128;
constexpr int HW_  = H_ * W_;
constexpr int TH   = 4;                        // output rows per block
constexpr int NTHR = 512;                      // 16 waves
constexpr float DECAY = 0.60653065971263342f;  // exp(-1/tau), tau = 2

// LDS input slab: [TH+2 rows][130 cols (w = -1..128)][64 ci] as f16.
constexpr int LCOLS = 130;
constexpr int XROWS = TH + 2;                  // 6
constexpr int XS_HALVES = XROWS * LCOLS * CIN; // 49920 halves = 99,840 B
constexpr int WF_HALVES = 18 * 4 * 32 * 16;    // 36864 halves = 73,728 B

// ---------------------------------------------------------------------------
// Weight pre-swizzle: fp32 [co][ci][3][3] -> f16 B-fragments in workspace.
// K order: k = (dh*3+dw)*64 + ci, split into 18 chunks of 32.
// B fragment (32x16, 16-bit, wave32): lane l: n = l&15, K = (l>>4)*8 + j for
// j < 8, K = 16 + (l>>4)*8 + (j-8) for j >= 8 -> 32 B per lane per fragment.
// ---------------------------------------------------------------------------
__global__ void snn_prep_wfrag(const float* __restrict__ w,
                               _Float16* __restrict__ wf) {
    int idx = blockIdx.x * 256 + threadIdx.x;       // 0 .. 18*4*32-1
    if (idx >= 18 * 4 * 32) return;
    int lane = idx & 31;
    int nt   = (idx >> 5) & 3;
    int kc   = idx >> 7;                            // 0..17
    int dhdw = kc >> 1;
    int dh   = dhdw / 3, dw = dhdw % 3;
    int cib  = (kc & 1) * 32;
    int n    = lane & 15;
    int khalf = (lane >> 4) * 8;
    int co   = nt * 16 + n;
    _Float16* dst = wf + (size_t)idx * 16;
#pragma unroll
    for (int j = 0; j < 16; ++j) {
        int ci = cib + khalf + j + (j >= 8 ? 8 : 0);
        dst[j] = (_Float16)w[((co * CIN + ci) * 3 + dh) * 3 + dw];
    }
}

// A fragment (16x32 f16): lane l -> M = l&15; with K base mhi per half-wave,
// halves 0..7 = K+0..7 and halves 8..15 = K+16..23 -> two 16B ds_load_b128.
__device__ inline v16h load_afrag(const _Float16* ap) {
    v8h lo = *(const v8h*)(ap);
    v8h hi = *(const v8h*)(ap + 16);
    return __builtin_shufflevector(lo, hi,
           0,1,2,3,4,5,6,7,8,9,10,11,12,13,14,15);
}

// ---------------------------------------------------------------------------
// Main fused kernel. Grid = B * H/TH blocks (one per (b, h0) 4-row stripe).
// 16 waves: output row rw = wv>>2, pixels w0 = (wv&3)*32 .. +31 (2 M-tiles),
// all 64 output channels (4 N-tiles).
// ---------------------------------------------------------------------------
__global__ __launch_bounds__(NTHR) void snn_conv_lif(
    const float*    __restrict__ x,     // [T][B][CIN][H][W] fp32
    const _Float16* __restrict__ wf,    // swizzled B fragments (global)
    const float*    __restrict__ bias,  // [COUT] fp32
    float*          __restrict__ out)   // [T][B][COUT][H][W] fp32 spikes
{
    __shared__ _Float16 ldsx[XS_HALVES];   // input slab (f16)
    __shared__ _Float16 ldsw[WF_HALVES];   // weight fragment table (f16)

    const int b    = blockIdx.x >> 5;          // batch
    const int h0   = (blockIdx.x & 31) * TH;   // first output row of stripe
    const int tid  = threadIdx.x;
    const int lane = tid & 31;
    const int wv   = tid >> 5;                 // wave 0..15
    const int rw   = wv >> 2;                  // output row within stripe
    const int w0   = (wv & 3) * 32;            // first pixel of wave's 2 tiles
    const int n    = lane & 15;                // N index within 16-wide tile
    const int mhi  = (lane >> 4) * 8;          // A K-offset / C-D row offset

    // ---- one-time: stage the 72 KB weight fragment table into LDS via TDM.
    // 1-D descriptor, 9216 x 8 B units, global -> LDS, no VGPR round-trip.
    // TDM ignores EXEC, so skip must be a *scalar* branch: readfirstlane(wv).
    if (__builtin_amdgcn_readfirstlane(wv) == 0) {
        const unsigned lds_addr = (unsigned)(uintptr_t)(void*)ldsw;
        const unsigned long long ga = (unsigned long long)(uintptr_t)wf;
        u32x4 g0;
        g0[0] = 1u;                               // count=1 valid descriptor
        g0[1] = lds_addr;                         // lds_addr (bytes)
        g0[2] = (unsigned)(ga & 0xFFFFFFFFu);     // global_addr[31:0]
        g0[3] = (unsigned)((ga >> 32) & 0x1FFFFFFu) | (2u << 30); // type=2
        i32x8 g1;
        g1[0] = (int)(3u << 16);                  // data_size=3 (8 B units)
        g1[1] = (int)(9216u << 16);               // tensor_dim0[15:0]
        g1[2] = (int)(1u << 16);                  // tensor_dim1 = 1
        g1[3] = (int)(9216u << 16);               // tile_dim0 = 9216
        g1[4] = 1;                                // tile_dim1 = 1
        g1[5] = 9216;                             // tensor_dim0_stride
        g1[6] = 0;
        g1[7] = 0;
        i32x4 gz4 = {0, 0, 0, 0};                 // groups 2/3 unused (<=2D)
        i32x8 gz8 = {0, 0, 0, 0, 0, 0, 0, 0};
        __builtin_amdgcn_tensor_load_to_lds(g0, g1, gz4, gz4, gz8, 0);
        __builtin_amdgcn_s_wait_tensorcnt(0);
    }

    // Per-lane bias for each of the 4 N-tiles (C layout: lane -> channel).
    float biasn[4];
#pragma unroll
    for (int nt = 0; nt < 4; ++nt) biasn[nt] = bias[nt * 16 + n];

    // Membrane state / accumulators: 2 M-tiles x 4 N-tiles x 8 rows = 64 VGPR.
    v8f acc[2][4];
#pragma unroll
    for (int s = 0; s < 2; ++s)
#pragma unroll
        for (int nt = 0; nt < 4; ++nt)
#pragma unroll
            for (int j = 0; j < 8; ++j) acc[s][nt][j] = 0.0f;

    for (int t = 0; t < T_; ++t) {
        __syncthreads();   // weights staged / prior timestep done with LDS
        // ---- stage x[t][b][:, h0-1..h0+4, -1..128] -> LDS f16 [r][col][ci]
        const float* xb = x + (((size_t)t * B_ + b) * CIN) * HW_;
        for (int i = tid; i < CIN * XROWS * LCOLS; i += NTHR) {
            int c   = i % LCOLS;               // col 0..129 -> w = c-1
            int rci = i / LCOLS;
            int r   = rci % XROWS;             // row 0..5 -> h0 + r - 1
            int ci  = rci / XROWS;
            int hh  = h0 + r - 1;
            int ww  = c - 1;
            float val = 0.0f;
            if ((unsigned)hh < (unsigned)H_ && (unsigned)ww < (unsigned)W_)
                val = xb[(size_t)ci * HW_ + hh * W_ + ww];
            ldsx[(r * LCOLS + c) * CIN + ci] = (_Float16)val;
        }
        __syncthreads();

        // ---- C = v*decay + bias rides the WMMA C operand
#pragma unroll
        for (int s = 0; s < 2; ++s)
#pragma unroll
            for (int nt = 0; nt < 4; ++nt)
#pragma unroll
                for (int j = 0; j < 8; ++j)
                    acc[s][nt][j] = acc[s][nt][j] * DECAY + biasn[nt];

        // ---- 18 K-chunks of 32: (dh,dw) x two halves of ci
        for (int kc = 0; kc < 18; ++kc) {
            const int dhdw = kc >> 1;
            const int dh   = dhdw / 3;
            const int dw   = dhdw % 3;
            const int cib  = (kc & 1) * 32;
            // input row for output row rw: (h0+rw) + dh - 1 -> slab row rw+dh
            const _Float16* ap =
                &ldsx[((rw + dh) * LCOLS + w0 + n + dw) * CIN + cib + mhi];
            v16h a0 = load_afrag(ap);             // M-tile 0 (w0 .. w0+15)
            v16h a1 = load_afrag(ap + 16 * CIN);  // M-tile 1 (w0+16 .. w0+31)
            // all 4 B fragments up front -> staggered waits, 8 WMMAs in a row
            const _Float16* bp = ldsw + (size_t)kc * 4 * 32 * 16 + lane * 16;
            v16h bf0 = *(const v16h*)(bp);
            v16h bf1 = *(const v16h*)(bp + 32 * 16);
            v16h bf2 = *(const v16h*)(bp + 2 * 32 * 16);
            v16h bf3 = *(const v16h*)(bp + 3 * 32 * 16);
            acc[0][0] = __builtin_amdgcn_wmma_f32_16x16x32_f16(
                false, a0, false, bf0, (short)0, acc[0][0], false, false);
            acc[1][0] = __builtin_amdgcn_wmma_f32_16x16x32_f16(
                false, a1, false, bf0, (short)0, acc[1][0], false, false);
            acc[0][1] = __builtin_amdgcn_wmma_f32_16x16x32_f16(
                false, a0, false, bf1, (short)0, acc[0][1], false, false);
            acc[1][1] = __builtin_amdgcn_wmma_f32_16x16x32_f16(
                false, a1, false, bf1, (short)0, acc[1][1], false, false);
            acc[0][2] = __builtin_amdgcn_wmma_f32_16x16x32_f16(
                false, a0, false, bf2, (short)0, acc[0][2], false, false);
            acc[1][2] = __builtin_amdgcn_wmma_f32_16x16x32_f16(
                false, a1, false, bf2, (short)0, acc[1][2], false, false);
            acc[0][3] = __builtin_amdgcn_wmma_f32_16x16x32_f16(
                false, a0, false, bf3, (short)0, acc[0][3], false, false);
            acc[1][3] = __builtin_amdgcn_wmma_f32_16x16x32_f16(
                false, a1, false, bf3, (short)0, acc[1][3], false, false);
        }

        // ---- LIF: threshold fire, emit spikes (vector b128 stores), reset
        float* ob = out + (((size_t)t * B_ + b) * COUT) * HW_
                        + (h0 + rw) * W_;
#pragma unroll
        for (int s = 0; s < 2; ++s) {
#pragma unroll
            for (int nt = 0; nt < 4; ++nt) {
                float* on = ob + (size_t)(nt * 16 + n) * HW_
                               + w0 + s * 16 + mhi;   // 8 consecutive floats
                float sp[8];
#pragma unroll
                for (int r = 0; r < 8; ++r) {
                    float vt = acc[s][nt][r];
                    sp[r] = vt > 1.0f ? 1.0f : 0.0f;
                    acc[s][nt][r] = (sp[r] > 0.0f) ? 0.0f : vt;  // hard reset to 0
                }
                *(float4*)(on)     = make_float4(sp[0], sp[1], sp[2], sp[3]);
                *(float4*)(on + 4) = make_float4(sp[4], sp[5], sp[6], sp[7]);
            }
        }
    }
}

extern "C" void kernel_launch(void* const* d_in, const int* in_sizes, int n_in,
                              void* d_out, int out_size, void* d_ws, size_t ws_size,
                              hipStream_t stream) {
    const float* x    = (const float*)d_in[0];   // [4,8,64,128,128]
    const float* w    = (const float*)d_in[1];   // [64,64,3,3]
    const float* bias = (const float*)d_in[2];   // [64]
    float* out        = (float*)d_out;           // [4,8,64,128,128]
    _Float16* wf      = (_Float16*)d_ws;         // 73,728 B fragment table

    snn_prep_wfrag<<<dim3(9), dim3(256), 0, stream>>>(w, wf);
    snn_conv_lif<<<dim3(B_ * (H_ / TH)), dim3(NTHR), 0, stream>>>(x, wf, bias, out);
}
